// Simple_BiLSTM_45148696215895
// MI455X (gfx1250) — matrix operand order repaired and checked by reading, not compiled, not run on hardware
//
#include <hip/hip_runtime.h>
#include <hip/hip_bf16.h>

typedef __attribute__((ext_vector_type(16))) _Float16 v16h;
typedef __attribute__((ext_vector_type(8)))  _Float16 v8h;
typedef __attribute__((ext_vector_type(8)))  float    v8f;

union V16H { v16h v; _Float16 h[16]; };
union V8F  { v8f  v; float    f[8];  };

// ---------------------------------------------------------------------------
// Fragment loaders for V_WMMA_F32_16X16X32_F16 (ISA 7.12.2 layouts)
// A (16x32, 16-bit): lane 0-15 -> row=lane, halfs [K0..7],[K16..23];
//                    lane 16-31 -> row=lane-16, halfs [K8..15],[K24..31]
// ---------------------------------------------------------------------------
__device__ __forceinline__ v16h load_a_frag(const _Float16* base, int ldk,
                                            int row, int k0, int hl) {
    const _Float16* p = base + (size_t)row * ldk + k0 + hl * 8;
    v8h lo = *(const v8h*)p;        // K = k0+hl*8   .. +7
    v8h hi = *(const v8h*)(p + 16); // K = k0+hl*8+16.. +23
    V16H a;
#pragma unroll
    for (int i = 0; i < 8; ++i) { a.h[i] = lo[i]; a.h[8 + i] = hi[i]; }
    return a.v;
}

// ---------------------------------------------------------------------------
// Pack a row-major f32 [K,N] matrix into WMMA B-fragment layout (f16):
// fragment f = kb*(Npad/16)+nt is 512 halfs: element (lane,i) = B[kb*32 +
// (lane>>4)*8 + (i&7) + (i>>3)*16][nt*16 + (lane&15)].  One 32B load per lane at use time.
// ---------------------------------------------------------------------------
__global__ void pack_b_kernel(const float* __restrict__ src, _Float16* __restrict__ dst,
                              int K, int N, int Npad) {
    int o = blockIdx.x * blockDim.x + threadIdx.x;
    if (o >= K * Npad) return;
    int f = o >> 9;
    int r = o & 511;
    int lane = r >> 4;
    int i = r & 15;
    int ntiles = Npad >> 4;
    int kb = f / ntiles;
    int nt = f - kb * ntiles;
    int k = kb * 32 + (lane >> 4) * 8 + (i & 7) + (i >> 3) * 16;
    int n = nt * 16 + (lane & 15);
    float v = (n < N && k < K) ? src[(size_t)k * N + n] : 0.0f;
    dst[o] = (_Float16)v;
}

// Embedding gather + f32->f16 cast. x[bt, e] = emb[tok[bt], e], bt = b*T+t.
__global__ void embed_cast_kernel(const int* __restrict__ tok,
                                  const float* __restrict__ emb,
                                  _Float16* __restrict__ x, int total) {
    int o = blockIdx.x * blockDim.x + threadIdx.x;
    if (o >= total) return;
    int bt = o >> 7;           // E = 128
    int e  = o & 127;
    x[o] = (_Float16)emb[(size_t)tok[bt] * 128 + e];
}

// ---------------------------------------------------------------------------
// WMMA GEMM: C[M,Npad] = A[M,K] (f16 row-major) @ packedB + bias, opt relu.
// Each wave computes a 16x64 strip (4 accumulator tiles sharing one A frag).
// ---------------------------------------------------------------------------
template <int ADD_BIAS, int RELU, int OUT_F16>
__global__ void wmma_gemm_kernel(const _Float16* __restrict__ A,
                                 const _Float16* __restrict__ pB,
                                 const float* __restrict__ bias,
                                 void* __restrict__ C,
                                 int M, int K, int Npad, int Nreal, int ldc) {
    int tid  = threadIdx.x;
    int lane = tid & 31;
    int wave = tid >> 5;
    int hl   = lane >> 4;
    int ln   = lane & 15;
    int nj   = Npad >> 6;                       // jobs along N (64-wide)
    int gw   = blockIdx.x * (blockDim.x >> 5) + wave;
    int rowTile = gw / nj;
    int colJob  = gw - rowTile * nj;
    if (rowTile >= (M >> 4)) return;
    int row0 = rowTile << 4;
    int n0   = colJob << 6;
    int nt0  = n0 >> 4;
    int ntiles = Npad >> 4;

    V8F acc[4];
#pragma unroll
    for (int j = 0; j < 4; ++j)
#pragma unroll
        for (int r = 0; r < 8; ++r) acc[j].f[r] = 0.0f;

    const v16h* pBv = (const v16h*)pB;
    for (int kb = 0; kb < (K >> 5); ++kb) {
        v16h a = load_a_frag(A, K, row0 + ln, kb * 32, hl);
#pragma unroll
        for (int j = 0; j < 4; ++j) {
            v16h b = pBv[(size_t)(kb * ntiles + nt0 + j) * 32 + lane];
            acc[j].v = __builtin_amdgcn_wmma_f32_16x16x32_f16(
                false, a, false, b, (short)0, acc[j].v, false, false);
        }
    }

#pragma unroll
    for (int j = 0; j < 4; ++j) {
        int col = n0 + j * 16 + ln;
        if (col < Nreal) {
#pragma unroll
            for (int r = 0; r < 8; ++r) {
                int row = row0 + hl * 8 + r;      // D layout: M = hl*8 + vgpr
                float v = acc[j].f[r];
                if (ADD_BIAS) v += bias[col];
                if (RELU)     v = fmaxf(v, 0.0f);
                if (OUT_F16)  ((_Float16*)C)[(size_t)row * ldc + col] = (_Float16)v;
                else          ((float*)C)[(size_t)row * ldc + col] = v;
            }
        }
    }
}

// ---------------------------------------------------------------------------
// Persistent LSTM recurrence. One block per direction, 512 threads = 16 waves.
// Wave w owns hidden cols [w*16, w*16+16) across all 4 gates -> gate math in
// registers. h double-buffered in LDS (f16), c in registers.
// Batch dim is processed in 2 phases of 32 rows so only acc[2][4] (64 VGPRs)
// is live at once. A uniform zero index laundered through an empty asm per
// phase stops the compiler hoisting/CSE-ing the 32 U-fragments (256 VGPRs)
// across phases/timesteps, while keeping the global address space of pU so
// the fragment loads lower to global_load_b128 (LOADcnt only, no DScnt).
// z = preact[t] + h @ U.  Gate order i,f,g,o (Keras, activation=relu):
//   c = sig(f)*c + sig(i)*relu(g);  h = sig(o)*relu(c)
// ---------------------------------------------------------------------------
__global__ void lstm_kernel(const _Float16* __restrict__ preA,  // dir 0 preacts [B*T,1024]
                            const _Float16* __restrict__ preB,  // dir 1 preacts
                            const _Float16* __restrict__ pUa,   // dir 0 packed U frags
                            const _Float16* __restrict__ pUb,   // dir 1 packed U frags
                            _Float16* __restrict__ hs_out,      // [B*T, hs_ld] or null
                            int hs_ld,
                            _Float16* __restrict__ h_last,      // [64,256] f16 or null
                            int T) {
    __shared__ __align__(32) _Float16 h_sh[2][64 * 256];  // ping-pong, 64 KB

    const int dir = blockIdx.x;
    const _Float16* pre = dir ? preB : preA;
    const v16h* pU = (const v16h*)(dir ? pUb : pUa);

    int tid  = threadIdx.x;
    int lane = tid & 31;
    int wave = tid >> 5;
    int hl   = lane >> 4;
    int ln   = lane & 15;
    int j0   = wave << 4;
    int j    = j0 + ln;

    for (int i = tid; i < 64 * 256; i += blockDim.x) h_sh[0][i] = (_Float16)0.0f;

    float c_reg[4][8];
#pragma unroll
    for (int mt = 0; mt < 4; ++mt)
#pragma unroll
        for (int r = 0; r < 8; ++r) c_reg[mt][r] = 0.0f;

    __syncthreads();

    for (int t = 0; t < T; ++t) {
        int tt = dir ? (T - 1 - t) : t;
        const _Float16* hr = h_sh[t & 1];        // read previous h
        _Float16*       hw = h_sh[(t + 1) & 1];  // write new h

#pragma unroll
        for (int p = 0; p < 2; ++p) {            // batch rows p*32 .. p*32+31
            // Opaque (runtime-zero) uniform index: blocks LICM/CSE of the U
            // fragment loads across phases/timesteps without erasing the
            // global address space of pU.
            unsigned uoff = 0;
            asm volatile("" : "+s"(uoff));

            V8F acc[2][4];                       // [m-tile][gate]
#pragma unroll
            for (int mi = 0; mi < 2; ++mi)
#pragma unroll
                for (int g = 0; g < 4; ++g)
#pragma unroll
                    for (int r = 0; r < 8; ++r) acc[mi][g].f[r] = 0.0f;

            for (int kb = 0; kb < 8; ++kb) {
                v16h a[2];
#pragma unroll
                for (int mi = 0; mi < 2; ++mi)
                    a[mi] = load_a_frag(hr, 256, (p * 2 + mi) * 16 + ln,
                                        kb * 32, hl);
                v16h bf[4];
#pragma unroll
                for (int g = 0; g < 4; ++g)
                    bf[g] = pU[(size_t)(kb * 64 + g * 16 + wave) * 32 + lane +
                               uoff];
#pragma unroll
                for (int mi = 0; mi < 2; ++mi)
#pragma unroll
                    for (int g = 0; g < 4; ++g)
                        acc[mi][g].v = __builtin_amdgcn_wmma_f32_16x16x32_f16(
                            false, a[mi], false, bf[g], (short)0,
                            acc[mi][g].v, false, false);
            }

            // epilogue: z = acc + preact; gates; update c; write h
#pragma unroll
            for (int mi = 0; mi < 2; ++mi)
#pragma unroll
                for (int r = 0; r < 8; ++r) {
                    int mt = p * 2 + mi;
                    int b  = mt * 16 + hl * 8 + r;
                    size_t pb = ((size_t)b * T + tt) * 1024 + j;
                    float zi = acc[mi][0].f[r] + (float)pre[pb];
                    float zf = acc[mi][1].f[r] + (float)pre[pb + 256];
                    float zg = acc[mi][2].f[r] + (float)pre[pb + 512];
                    float zo = acc[mi][3].f[r] + (float)pre[pb + 768];
                    float si = 1.0f / (1.0f + __expf(-zi));
                    float sf = 1.0f / (1.0f + __expf(-zf));
                    float so = 1.0f / (1.0f + __expf(-zo));
                    float c  = sf * c_reg[mt][r] + si * fmaxf(zg, 0.0f);
                    c_reg[mt][r] = c;
                    float h  = so * fmaxf(c, 0.0f);
                    _Float16 hh = (_Float16)h;
                    hw[b * 256 + j] = hh;
                    if (hs_out)
                        hs_out[((size_t)b * T + tt) * hs_ld + dir * 256 + j] = hh;
                    if (h_last && t == T - 1)
                        h_last[b * 256 + j] = hh;
                }
        }

        __syncthreads();  // new h fully written before next step reads it
    }
}

// Row-wise softmax over Nreal columns (ld-padded input).
__global__ void softmax_kernel(const float* __restrict__ logits,
                               float* __restrict__ out, int ldl, int n) {
    __shared__ float red[256];
    int row = blockIdx.x;
    int tid = threadIdx.x;
    const float* x = logits + (size_t)row * ldl;

    float m = -1e30f;
    for (int i = tid; i < n; i += 256) m = fmaxf(m, x[i]);
    red[tid] = m;
    __syncthreads();
    for (int s = 128; s > 0; s >>= 1) {
        if (tid < s) red[tid] = fmaxf(red[tid], red[tid + s]);
        __syncthreads();
    }
    m = red[0];
    __syncthreads();

    float sum = 0.0f;
    for (int i = tid; i < n; i += 256) sum += __expf(x[i] - m);
    red[tid] = sum;
    __syncthreads();
    for (int s = 128; s > 0; s >>= 1) {
        if (tid < s) red[tid] += red[tid + s];
        __syncthreads();
    }
    float inv = 1.0f / red[0];
    for (int i = tid; i < n; i += 256)
        out[(size_t)row * n + i] = __expf(x[i] - m) * inv;
}

// ---------------------------------------------------------------------------
extern "C" void kernel_launch(void* const* d_in, const int* in_sizes, int n_in,
                              void* d_out, int out_size, void* d_ws, size_t ws_size,
                              hipStream_t stream) {
    (void)in_sizes; (void)n_in; (void)out_size; (void)ws_size;

    const int*   tokens = (const int*)d_in[0];
    const float* emb    = (const float*)d_in[1];
    const float* W1f    = (const float*)d_in[2];
    const float* U1f    = (const float*)d_in[3];
    const float* b1f    = (const float*)d_in[4];
    const float* W1b    = (const float*)d_in[5];
    const float* U1b    = (const float*)d_in[6];
    const float* b1b    = (const float*)d_in[7];
    const float* W2     = (const float*)d_in[8];
    const float* U2m    = (const float*)d_in[9];
    const float* b2     = (const float*)d_in[10];
    const float* W3     = (const float*)d_in[11];
    const float* b3     = (const float*)d_in[12];
    const float* W4     = (const float*)d_in[13];
    const float* b4     = (const float*)d_in[14];

    const int B = 64, T = 256, E = 128, M = B * T;       // M = 16384
    const int H4 = 1024, U3 = 512;
    const int N4real = 5000, N4pad = 5056;               // 5056 = 79*64

    char* ws = (char*)d_ws;
    size_t off = 0;
    auto alloc = [&](size_t bytes) {
        size_t o = off;
        off += (bytes + 255) & ~(size_t)255;
        return o;
    };

    size_t o_xh    = alloc((size_t)M * E * 2);            // x  f16 [M,128]
    size_t o_pW1f  = alloc((size_t)E * H4 * 2);
    size_t o_pW1b  = alloc((size_t)E * H4 * 2);
    size_t o_pU1f  = alloc((size_t)256 * H4 * 2);
    size_t o_pU1b  = alloc((size_t)256 * H4 * 2);
    size_t o_pW2   = alloc((size_t)512 * H4 * 2);
    size_t o_pU2   = alloc((size_t)256 * H4 * 2);
    size_t o_pW3   = alloc((size_t)256 * U3 * 2);
    size_t o_pW4   = alloc((size_t)512 * N4pad * 2);
    size_t o_x2h   = alloc((size_t)M * 512 * 2);          // concat hs f16
    size_t o_pre1f = alloc((size_t)M * H4 * 2);           // preacts f16
    size_t o_pre1b = alloc((size_t)M * H4 * 2);
    size_t o_pre2  = o_pre1f;                             // reuse after layer 1
    size_t o_hl    = alloc((size_t)B * 256 * 2);          // last h f16
    size_t o_h3    = alloc((size_t)B * U3 * 2);           // relu dense f16
    size_t o_log   = alloc((size_t)B * N4pad * 4);        // logits f32 padded

    auto packed = [&](size_t o) { return (_Float16*)(ws + o); };

    // 1) pack all weight matrices into WMMA B-fragment layout
    auto pack = [&](const float* src, size_t dsto, int K, int N, int Npad) {
        int total = K * Npad;
        pack_b_kernel<<<(total + 255) / 256, 256, 0, stream>>>(
            src, packed(dsto), K, N, Npad);
    };
    pack(W1f, o_pW1f, 128, H4, H4);
    pack(W1b, o_pW1b, 128, H4, H4);
    pack(U1f, o_pU1f, 256, H4, H4);
    pack(U1b, o_pU1b, 256, H4, H4);
    pack(W2,  o_pW2,  512, H4, H4);
    pack(U2m, o_pU2,  256, H4, H4);
    pack(W3,  o_pW3,  256, U3, U3);
    pack(W4,  o_pW4,  512, N4real, N4pad);

    // 2) embedding gather + cast
    {
        int total = M * E;
        embed_cast_kernel<<<(total + 255) / 256, 256, 0, stream>>>(
            tokens, emb, packed(o_xh), total);
    }

    auto gemm_blocks = [](int M_, int Npad_) {
        int waves = (M_ >> 4) * (Npad_ >> 6);
        return (waves + 7) / 8;
    };

    // 3) time-parallel input projections: preact1{f,b} = x @ W1{f,b} + b
    wmma_gemm_kernel<1, 0, 1><<<gemm_blocks(M, H4), 256, 0, stream>>>(
        packed(o_xh), packed(o_pW1f), b1f, ws + o_pre1f, M, 128, H4, H4, H4);
    wmma_gemm_kernel<1, 0, 1><<<gemm_blocks(M, H4), 256, 0, stream>>>(
        packed(o_xh), packed(o_pW1b), b1b, ws + o_pre1b, M, 128, H4, H4, H4);

    // 4) BiLSTM layer 1: 2 persistent workgroups (fwd / bwd), write concat hs
    lstm_kernel<<<2, 512, 0, stream>>>(
        packed(o_pre1f), packed(o_pre1b), packed(o_pU1f), packed(o_pU1b),
        packed(o_x2h), 512, (_Float16*)nullptr, T);

    // 5) preact2 = x2 @ W2 + b2   (reuses preact1f storage)
    wmma_gemm_kernel<1, 0, 1><<<gemm_blocks(M, H4), 256, 0, stream>>>(
        packed(o_x2h), packed(o_pW2), b2, ws + o_pre2, M, 512, H4, H4, H4);

    // 6) LSTM layer 2: single persistent workgroup, keep only last h
    lstm_kernel<<<1, 512, 0, stream>>>(
        packed(o_pre2), packed(o_pre2), packed(o_pU2), packed(o_pU2),
        (_Float16*)nullptr, 0, packed(o_hl), T);

    // 7) h3 = relu(h_last @ W3 + b3)  (f16 out, fused epilogue)
    wmma_gemm_kernel<1, 1, 1><<<gemm_blocks(B, U3), 256, 0, stream>>>(
        packed(o_hl), packed(o_pW3), b3, ws + o_h3, B, 256, U3, U3, U3);

    // 8) logits = h3 @ W4 + b4  (f32, N padded to 5056, guarded stores)
    wmma_gemm_kernel<1, 0, 0><<<gemm_blocks(B, N4pad), 256, 0, stream>>>(
        packed(o_h3), packed(o_pW4), b4, ws + o_log, B, 512, N4pad, N4real,
        N4pad);

    // 9) softmax rows -> d_out [64, 5000] f32
    softmax_kernel<<<B, 256, 0, stream>>>((const float*)(ws + o_log),
                                          (float*)d_out, N4pad, N4real);
}
